// GenerationIntervalAttention_32830730010767
// MI455X (gfx1250) — matrix operand order repaired and checked by reading, not compiled, not run on hardware
//
#include <hip/hip_runtime.h>

typedef __attribute__((ext_vector_type(16))) _Float16 v16h;
typedef __attribute__((ext_vector_type(8)))  _Float16 v8h;
typedef __attribute__((ext_vector_type(8)))  float    v8f;
typedef __attribute__((ext_vector_type(8)))  unsigned  v8u;
typedef __attribute__((ext_vector_type(4)))  unsigned  v4u;

namespace {
constexpr int kT = 64;      // sequence length
constexpr int kD = 32;      // model dim
constexpr int kH = 4;       // heads (HD = 8)
constexpr int XS = kD + 2;  // padded LDS row stride (halves) for [T][D] tiles
constexpr int WS = kD + 2;  // padded stride for wT [n][k]
constexpr int VS = kT + 8;  // padded stride for v^T [D][T] (144B rows, 16B aligned)
constexpr int AS = kT + 8;  // padded stride for attn [T][T] (144B rows, 16B aligned)
}

__device__ inline unsigned pack2h(float a, float b) {
  union { unsigned u; _Float16 h[2]; } t;
  t.h[0] = (_Float16)a;
  t.h[1] = (_Float16)b;
  return t.u;
}

__device__ inline void unpack2h(unsigned u, float& a, float& b) {
  union { unsigned u; _Float16 h[2]; } t;
  t.u = u;
  a = (float)t.h[0];
  b = (float)t.h[1];
}

__device__ inline v16h as_v16h(v8u u) {
  union { v8u u; v16h h; } t;
  t.u = u;
  return t.h;
}

__device__ inline v4u as_v4u(v8h h) {
  union { v8h h; v4u u; } t;
  t.h = h;
  return t.u;
}

__device__ inline v8f splat8(float x) {
  v8f r;
#pragma unroll
  for (int i = 0; i < 8; ++i) r[i] = x;
  return r;
}

__device__ inline v8h splat8h(_Float16 x) {
  v8h r;
#pragma unroll
  for (int i = 0; i < 8; ++i) r[i] = x;
  return r;
}

__global__ __launch_bounds__(128)
void gia_attention_kernel(const float* __restrict__ x,
                          const float* __restrict__ q_w, const float* __restrict__ q_b,
                          const float* __restrict__ k_w, const float* __restrict__ k_b,
                          const float* __restrict__ v_w, const float* __restrict__ v_b,
                          const float* __restrict__ out_w, const float* __restrict__ out_b,
                          const float* __restrict__ rel, const float* __restrict__ log_gi,
                          float* __restrict__ out) {
  __shared__ __align__(16) _Float16 xh[kT * XS];      // x  (f16), [t][d]
  __shared__ __align__(16) _Float16 qh[kT * XS];      // q  (f16), [t][d]
  __shared__ __align__(16) _Float16 kh[kT * XS];      // k  (f16), [t][d]
  __shared__ __align__(16) _Float16 vT[kD * VS];      // v^T (f16), [d][t]
  __shared__ __align__(16) _Float16 oh[kT * XS];      // attn out pre-proj, [t][d]
  __shared__ __align__(16) _Float16 ah[kH][kT * AS];  // per-head scores -> probs
  __shared__ __align__(16) _Float16 wT[4][kD * WS];   // weights^T f16: [n][k]
  __shared__ float gimt[kT];                          // gi mask LUT over lag

  const int tid   = threadIdx.x;
  const int wave  = tid >> 5;     // 0..3  (head id / row-tile id)
  const int lane  = tid & 31;
  const int lh    = lane & 15;    // lane % 16
  const int lhalf = lane >> 4;    // 0 or 1
  const int bn    = blockIdx.x;

  const float* xp = x + (size_t)bn * (kT * kD);

  // ---------------- Phase 0: stage x + transposed f16 weights + LUT -----
#pragma unroll
  for (int p = 0; p < 8; ++p) {
    const int g = (tid + 128 * p) << 1;  // even element index
    const float a = xp[g], b = xp[g + 1];
    *(unsigned*)&xh[(g >> 5) * XS + (g & 31)] = pack2h(a, b);
  }
  // weights: wT[mat][n][k], packed pairs over k (rows 2p, 2p+1 of w)
#pragma unroll
  for (int matw = 0; matw < 4; ++matw) {
    const float* w = (matw == 0) ? q_w : (matw == 1) ? k_w : (matw == 2) ? v_w : out_w;
#pragma unroll
    for (int it = 0; it < 4; ++it) {
      const int idx = tid + 128 * it;   // 0..511
      const int n = idx & 31;           // coalesced over lanes
      const int p = idx >> 5;           // k-pair 0..15
      const float a = w[(2 * p) * kD + n];
      const float b = w[(2 * p + 1) * kD + n];
      *(unsigned*)&wT[matw][n * WS + 2 * p] = pack2h(a, b);
    }
  }
  if (tid < kT) {
    const float gen = __expf(log_gi[0]);
    const float lag = (float)tid;
    const float z   = (lag - gen) / (0.5f * gen);
    gimt[tid] = 0.7f * __expf(-0.5f * z * z) + 0.3f * __expf(-0.1f * lag);
  }
  __syncthreads();

  // ---------------- Phase 1: Q/K/V projections (24 WMMA tiles) ----------
  // Fully unrolled: s -> (mat = s/2, nt = s&1) are compile-time; mt = wave.
  {
    // A tile (x rows 16*wave..+15) is shared by all 6 WMMAs of this wave.
    v8u au;
    const int m = 16 * wave + lh;
#pragma unroll
    for (int j = 0; j < 8; ++j) {
      const int kb = ((j >> 2) << 4) + (lhalf << 3) + ((j & 3) << 1);
      au[j] = *(const unsigned*)&xh[m * XS + kb];
    }
    const v16h av = as_v16h(au);

#pragma unroll
    for (int s = 0; s < 6; ++s) {
      const int mat = s >> 1;                   // 0=q 1=k 2=v (constant)
      const int nt  = s & 1;                    // constant
      const float* bias = (mat == 0) ? q_b : (mat == 1) ? k_b : v_b;

      v8u bu;
      const int n = 16 * nt + lh;
#pragma unroll
      for (int j = 0; j < 8; ++j) {
        const int kb = ((j >> 2) << 4) + (lhalf << 3) + ((j & 3) << 1);
        bu[j] = *(const unsigned*)&wT[mat][n * WS + kb];
      }

      // f16 accumulate/result: D half i <-> row M = 16*wave + i + 8*lhalf
      v8h ch = splat8h((_Float16)bias[n]);
      v8h dh = __builtin_amdgcn_wmma_f16_16x16x32_f16(
          false, av, false, as_v16h(bu), (short)0, ch, false, false);
      if (mat == 2) {
        *(v4u*)&vT[n * VS + 16 * wave + 8 * lhalf] = as_v4u(dh);  // packed b128
      } else {
        _Float16* dst = (mat == 0) ? qh : kh;
#pragma unroll
        for (int i = 0; i < 8; ++i)
          dst[(16 * wave + i + 8 * lhalf) * XS + n] = dh[i];
      }
    }
  }
  __syncthreads();

  // ---------------- Phase 2: scores = (q k^T)/sqrt(hd) + rel, * gi ------
  const int h = wave;
  _Float16* ahh = ah[h];
  const float inv_scale = 0.35355339059327373f;  // 1/sqrt(8)

  for (int mt = 0; mt < 4; ++mt) {
    for (int ntk = 0; ntk < 4; ++ntk) {
      // A: q rows, K window [8h, 8h+8) kept, rest zeroed
      v8u au;
      const int m = 16 * mt + lh;
#pragma unroll
      for (int j = 0; j < 8; ++j) {
        const int kb = ((j >> 2) << 4) + (lhalf << 3) + ((j & 3) << 1);
        au[j] = ((kb >> 3) == h) ? *(const unsigned*)&qh[m * XS + kb] : 0u;
      }
      // B: k^T -> columns are key positions; same K-window masking
      v8u bu;
      const int tk = 16 * ntk + lh;
#pragma unroll
      for (int j = 0; j < 8; ++j) {
        const int kb = ((j >> 2) << 4) + (lhalf << 3) + ((j & 3) << 1);
        bu[j] = ((kb >> 3) == h) ? *(const unsigned*)&kh[tk * XS + kb] : 0u;
      }
      v8f c = splat8(0.0f);
      v8f d = __builtin_amdgcn_wmma_f32_16x16x32_f16(
          false, as_v16h(au), false, as_v16h(bu), (short)0, c, false, false);
#pragma unroll
      for (int i = 0; i < 8; ++i) {
        const int tq  = 16 * mt + i + 8 * lhalf;
        const int tkk = 16 * ntk + lh;
        const int lag = tq - tkk;
        const float s  = d[i] * inv_scale + rel[h * (kT * kT) + tq * kT + tkk];
        const float gm = (lag >= 0) ? gimt[lag] : 0.0f;
        ahh[tq * AS + tkk] = (_Float16)(s * gm);
      }
    }
  }
  __syncthreads();

  // ---------------- Phase 3: row softmax, vectorized b128 LDS -----------
  for (int r = lane; r < kT; r += 32) {
    v4u* rp = (v4u*)&ahh[r * AS];  // 144B row base, 16B aligned
    v4u buf[8];
#pragma unroll
    for (int cgi = 0; cgi < 8; ++cgi) buf[cgi] = rp[cgi];

    float vals[kT];
#pragma unroll
    for (int cgi = 0; cgi < 8; ++cgi)
#pragma unroll
      for (int e = 0; e < 4; ++e)
        unpack2h(buf[cgi][e], vals[cgi * 8 + e * 2], vals[cgi * 8 + e * 2 + 1]);

    float mx = -3.0e38f;
#pragma unroll
    for (int i = 0; i < kT; ++i) mx = fmaxf(mx, vals[i]);
    float sum = 0.0f;
#pragma unroll
    for (int i = 0; i < kT; ++i) { vals[i] = __expf(vals[i] - mx); sum += vals[i]; }
    const float inv = 1.0f / sum;

#pragma unroll
    for (int cgi = 0; cgi < 8; ++cgi) {
      v4u ov;
#pragma unroll
      for (int e = 0; e < 4; ++e)
        ov[e] = pack2h(vals[cgi * 8 + e * 2] * inv, vals[cgi * 8 + e * 2 + 1] * inv);
      rp[cgi] = ov;
    }
  }
  __syncthreads();

  // ---------------- Phase 4: out_h = P @ v_h  (N padded 8 -> 16) --------
  for (int mt = 0; mt < 4; ++mt) {
    v8h acch = splat8h((_Float16)0.0f);
#pragma unroll
    for (int kt = 0; kt < 2; ++kt) {
      v8u au;
      const int m = 16 * mt + lh;
#pragma unroll
      for (int j = 0; j < 8; ++j) {
        const int kb = 32 * kt + ((j >> 2) << 4) + (lhalf << 3) + ((j & 3) << 1);
        au[j] = *(const unsigned*)&ahh[m * AS + kb];
      }
      v8u bu;
      const int vcol = 8 * h + lh;  // valid only for lh < 8
#pragma unroll
      for (int j = 0; j < 8; ++j) {
        const int kb = 32 * kt + ((j >> 2) << 4) + (lhalf << 3) + ((j & 3) << 1);
        bu[j] = (lh < 8) ? *(const unsigned*)&vT[vcol * VS + kb] : 0u;
      }
      acch = __builtin_amdgcn_wmma_f16_16x16x32_f16(
          false, as_v16h(au), false, as_v16h(bu), (short)0, acch, false, false);
    }
#pragma unroll
    for (int i = 0; i < 8; ++i) {
      const int tq = 16 * mt + i + 8 * lhalf;
      if (lh < 8) oh[tq * XS + 8 * h + lh] = acch[i];
    }
  }
  __syncthreads();

  // ---------------- Phase 5: final projection, f32 out ------------------
  {
    const int mt = wave;  // each wave does one 16-row tile, both col tiles
    const int m  = 16 * mt + lh;
    v8u au;
#pragma unroll
    for (int j = 0; j < 8; ++j) {
      const int kb = ((j >> 2) << 4) + (lhalf << 3) + ((j & 3) << 1);
      au[j] = *(const unsigned*)&oh[m * XS + kb];
    }
    const v16h av = as_v16h(au);
    float* op = out + (size_t)bn * (kT * kD);
#pragma unroll
    for (int nt = 0; nt < 2; ++nt) {
      const int n = 16 * nt + lh;
      v8u bu;
#pragma unroll
      for (int j = 0; j < 8; ++j) {
        const int kb = ((j >> 2) << 4) + (lhalf << 3) + ((j & 3) << 1);
        bu[j] = *(const unsigned*)&wT[3][n * WS + kb];
      }
      v8f c = splat8(out_b[n]);
      v8f d = __builtin_amdgcn_wmma_f32_16x16x32_f16(
          false, av, false, as_v16h(bu), (short)0, c, false, false);
#pragma unroll
      for (int i = 0; i < 8; ++i) {
        const int mm = 16 * mt + i + 8 * lhalf;
        op[mm * kD + n] = d[i];
      }
    }
  }
}

extern "C" void kernel_launch(void* const* d_in, const int* in_sizes, int n_in,
                              void* d_out, int out_size, void* d_ws, size_t ws_size,
                              hipStream_t stream) {
  const float* x      = (const float*)d_in[0];
  const float* q_w    = (const float*)d_in[1];
  const float* q_b    = (const float*)d_in[2];
  const float* k_w    = (const float*)d_in[3];
  const float* k_b    = (const float*)d_in[4];
  const float* v_w    = (const float*)d_in[5];
  const float* v_b    = (const float*)d_in[6];
  const float* out_w  = (const float*)d_in[7];
  const float* out_b  = (const float*)d_in[8];
  const float* rel    = (const float*)d_in[9];
  const float* log_gi = (const float*)d_in[10];
  float* out = (float*)d_out;

  const int BN = in_sizes[0] / (kT * kD);  // 8000
  gia_attention_kernel<<<BN, 128, 0, stream>>>(
      x, q_w, q_b, k_w, k_b, v_w, v_b, out_w, out_b, rel, log_gi, out);
}
